// RelationalGraphLayer_42494406427175
// MI455X (gfx1250) — matrix operand order repaired and checked by reading, compiled, not run on hardware
//
#include <hip/hip_runtime.h>
#include <hip/hip_bf16.h>

// ---------------------------------------------------------------------------
// Types for CDNA5 WMMA (wave32): bf16 A/B fragments (8 VGPRs), f32 C/D (8 VGPRs)
// ---------------------------------------------------------------------------
typedef __attribute__((ext_vector_type(16))) __bf16 bf16x16;
typedef __attribute__((ext_vector_type(8)))  __bf16 bf16x8;
typedef __attribute__((ext_vector_type(8)))  float  f32x8;

#define DEV static __device__ __forceinline__

DEV void atomicMaxF(float* addr, float val) {
  // sign-split trick: works for any mix of signs, init to -inf
  if (val >= 0.0f) atomicMax((int*)addr, __float_as_int(val));
  else             atomicMin((unsigned int*)addr, __float_as_uint(val));
}

DEV bf16x16 load_frag32B(const __bf16* p) {
  bf16x8 lo = *(const bf16x8*)(p);
  bf16x8 hi = *(const bf16x8*)(p + 8);
  return __builtin_shufflevector(lo, hi, 0,1,2,3,4,5,6,7,8,9,10,11,12,13,14,15);
}

// ---------------------------------------------------------------------------
// Fill
// ---------------------------------------------------------------------------
__global__ void k_fill(float* __restrict__ p, float v, long long n) {
  long long i = (long long)blockIdx.x * blockDim.x + threadIdx.x;
  if (i < n) p[i] = v;
}

// ---------------------------------------------------------------------------
// LayerNorm (fp32 in) -> bf16 out. One wave per 128-wide row.
// ---------------------------------------------------------------------------
__global__ void k_layernorm_bf16(const float* __restrict__ x,
                                 const float* __restrict__ g,
                                 const float* __restrict__ b,
                                 __bf16* __restrict__ y, int nrows) {
  int wave = (blockIdx.x * blockDim.x + threadIdx.x) >> 5;
  int lane = threadIdx.x & 31;
  if (wave >= nrows) return;
  const float* row = x + (long long)wave * 128;
  float4 v = *(const float4*)(row + lane * 4);
  float s  = v.x + v.y + v.z + v.w;
  float ss = v.x * v.x + v.y * v.y + v.z * v.z + v.w * v.w;
#pragma unroll
  for (int m = 1; m < 32; m <<= 1) {
    s  += __shfl_xor(s,  m, 32);
    ss += __shfl_xor(ss, m, 32);
  }
  float mu  = s * (1.0f / 128.0f);
  float var = ss * (1.0f / 128.0f) - mu * mu;
  float rs  = rsqrtf(var + 1e-5f);
  float4 gg = *(const float4*)(g + lane * 4);
  float4 bb = *(const float4*)(b + lane * 4);
  __bf16* yr = y + (long long)wave * 128 + lane * 4;
  yr[0] = (__bf16)((v.x - mu) * rs * gg.x + bb.x);
  yr[1] = (__bf16)((v.y - mu) * rs * gg.y + bb.y);
  yr[2] = (__bf16)((v.z - mu) * rs * gg.z + bb.z);
  yr[3] = (__bf16)((v.w - mu) * rs * gg.w + bb.w);
}

// ---------------------------------------------------------------------------
// Pack fp32 weight [128, N] (row-major) into bf16 WMMA B-fragments.
// Fragment (t = n/16, s = k/32) holds 32x16 bf16; lane = ((k>>3)&1)*16 + (n&15);
// half within lane = 8*((k>>4)&1) + (k&7).   (ISA 7.12.2 16-bit B layout)
// ---------------------------------------------------------------------------
__global__ void k_pack_weight(const float* __restrict__ W, __bf16* __restrict__ Bp, int N) {
  int i = blockIdx.x * blockDim.x + threadIdx.x;  // over 128*N elements
  if (i >= 128 * N) return;
  int k = i / N, n = i % N;
  int t = n >> 4, s = k >> 5;
  int g = (k >> 3) & 1, j = ((k >> 4) & 1) * 4 + ((k & 7) >> 1), h = k & 1;
  int lane = g * 16 + (n & 15);
  long long off = ((long long)(t * 4 + s)) * 512 + lane * 16 + j * 2 + h;
  Bp[off] = (__bf16)W[i];
}

// ---------------------------------------------------------------------------
// GEMM: C[M,N] = A[M,128] @ W[128,N] + bias, A bf16 row-major, W pre-packed.
// One wave -> 32 rows x 64 cols: two independent accumulator chains share the
// B fragments; interleaved WMMAs cover the D->C hazard gap with real work.
// B fragments double-buffered; bias preloaded so no loads interrupt the WMMA
// stream (keeps loadcnt waits partial). N compile-time -> immediate-offset
// stores. __launch_bounds__(256, 1) lifts the VGPR cap (~160 live regs) so
// nothing spills to scratch. M % 16 == 0 always; odd row-tile count -> has2.
// ---------------------------------------------------------------------------
template <int N>
__global__ void __launch_bounds__(256, 1)
k_gemm_wmma_bf16(const __bf16* __restrict__ A,
                 const __bf16* __restrict__ Bp,
                 const float* __restrict__ bias,
                 float* __restrict__ C, int mt) {
  constexpr int NT4 = N >> 6;                 // # of 64-col tiles
  const int lane = threadIdx.x & 31;
  const int tn = blockIdx.x * (blockDim.x >> 5) + (threadIdx.x >> 5);
  if (tn >= NT4) return;
  const int rt0 = blockIdx.y * 2;
  const bool has2 = (rt0 + 1) < mt;           // wave-uniform
  const int row0 = rt0 << 4;
  const int row1 = has2 ? row0 + 16 : row0;   // clamp tail A loads
  const int g = lane >> 4;                    // half-wave group
  const int r = lane & 15;                    // row (A) / col (B,C) within tile

  // Bias preload: keep the tile loop free of new loads.
  float bv[4];
#pragma unroll
  for (int t4 = 0; t4 < 4; ++t4) bv[t4] = bias[((tn * 4 + t4) << 4) + r];

  // A fragments for both row tiles: 4 K-steps each.
  // (A rows are 128 halves; the two 16-byte chunks sit at +0 and +16 halves.)
  const __bf16* ap0 = A + (long long)(row0 + r) * 128 + g * 8;
  const __bf16* ap1 = A + (long long)(row1 + r) * 128 + g * 8;
  bf16x16 a0[4], a1[4];
#pragma unroll
  for (int s = 0; s < 4; ++s) {
    a0[s] = load_frag32B(ap0 + s * 32) /* second chunk read at +8; redone below */;
    a1[s] = load_frag32B(ap1 + s * 32);
  }
  // load_frag32B reads +0..7 and +8..15 halves; A needs +0..7 and +16..23.
  // Redo with the correct second chunk:
#pragma unroll
  for (int s = 0; s < 4; ++s) {
    bf16x8 lo0 = *(const bf16x8*)(ap0 + s * 32);
    bf16x8 hi0 = *(const bf16x8*)(ap0 + s * 32 + 16);
    a0[s] = __builtin_shufflevector(lo0, hi0, 0,1,2,3,4,5,6,7,8,9,10,11,12,13,14,15);
    bf16x8 lo1 = *(const bf16x8*)(ap1 + s * 32);
    bf16x8 hi1 = *(const bf16x8*)(ap1 + s * 32 + 16);
    a1[s] = __builtin_shufflevector(lo1, hi1, 0,1,2,3,4,5,6,7,8,9,10,11,12,13,14,15);
  }

  // B fragment double buffer: prime tile 0, overlap next-tile loads with WMMAs.
  const __bf16* bp0 = Bp + (long long)(tn * 4) * 2048 + lane * 16;
  bf16x16 bbuf[2][4];
#pragma unroll
  for (int s = 0; s < 4; ++s) bbuf[0][s] = load_frag32B(bp0 + s * 512);

#pragma unroll
  for (int t4 = 0; t4 < 4; ++t4) {
    const int t = tn * 4 + t4;
    if (t4 < 3) {
      const __bf16* bpn = bp0 + (long long)(t4 + 1) * 2048;
      __builtin_prefetch(bpn + 2048, 0, 3);
#pragma unroll
      for (int s = 0; s < 4; ++s) bbuf[(t4 + 1) & 1][s] = load_frag32B(bpn + s * 512);
    }
    const float bvt = bv[t4];
    f32x8 acc0 = {bvt, bvt, bvt, bvt, bvt, bvt, bvt, bvt};
    f32x8 acc1 = acc0;
#pragma unroll
    for (int s = 0; s < 4; ++s) {
      acc0 = __builtin_amdgcn_wmma_f32_16x16x32_bf16(false, a0[s], false, bbuf[t4 & 1][s],
                                                     (short)0, acc0, false, false);
      acc1 = __builtin_amdgcn_wmma_f32_16x16x32_bf16(false, a1[s], false, bbuf[t4 & 1][s],
                                                     (short)0, acc1, false, false);
    }
    // C tiles: VGPR rr -> row = rowBase + g*8 + rr, col = t*16 + r
    float* c0 = C + (long long)(row0 + g * 8) * N + (t << 4) + r;
#pragma unroll
    for (int rr = 0; rr < 8; ++rr) c0[rr * N] = acc0[rr];
    if (has2) {
      float* c1 = C + (long long)(row0 + 16 + g * 8) * N + (t << 4) + r;
#pragma unroll
      for (int rr = 0; rr < 8; ++rr) c1[rr * N] = acc1[rr];
    }
  }
}

// ---------------------------------------------------------------------------
// Per-edge q.k logits: one wave per edge; 8 lanes per head (128 elems/head).
// ---------------------------------------------------------------------------
__global__ void k_edge_logits(const float* __restrict__ q, const float* __restrict__ k,
                              const int* __restrict__ src, const int* __restrict__ dst,
                              float* __restrict__ logit, int E) {
  int wave = (blockIdx.x * blockDim.x + threadIdx.x) >> 5;
  int lane = threadIdx.x & 31;
  if (wave >= E) return;
  int sn = src[wave], dn = dst[wave];
  int head = lane >> 3, sub = lane & 7;
  const float4* qp = (const float4*)(q + (long long)dn * 512 + head * 128 + sub * 16);
  const float4* kp = (const float4*)(k + (long long)sn * 512 + head * 128 + sub * 16);
  float acc = 0.f;
#pragma unroll
  for (int i = 0; i < 4; ++i) {
    float4 a = qp[i], b = kp[i];
    acc += a.x * b.x + a.y * b.y + a.z * b.z + a.w * b.w;
  }
  acc += __shfl_xor(acc, 1, 32);
  acc += __shfl_xor(acc, 2, 32);
  acc += __shfl_xor(acc, 4, 32);
  if (sub == 0) logit[(long long)wave * 4 + head] = acc * 0.08838834764831845f; // 1/sqrt(128)
}

__global__ void k_edge_max(const float* __restrict__ logit, const int* __restrict__ dst,
                           float* __restrict__ m, int E) {
  int i = blockIdx.x * blockDim.x + threadIdx.x;
  if (i >= E * 4) return;
  int e = i >> 2, h = i & 3;
  atomicMaxF(&m[(long long)dst[e] * 4 + h], logit[i]);
}

__global__ void k_edge_exp(const float* __restrict__ logit, const int* __restrict__ dst,
                           const float* __restrict__ m, float* __restrict__ p,
                           float* __restrict__ s, int E) {
  int i = blockIdx.x * blockDim.x + threadIdx.x;
  if (i >= E * 4) return;
  int e = i >> 2, h = i & 3;
  float pe = __expf(logit[i] - m[(long long)dst[e] * 4 + h]);
  p[i] = pe;
  atomicAdd(&s[(long long)dst[e] * 4 + h], pe);
}

// One wave per (edge, head): alpha * v scatter-add (128 floats, 4/lane).
__global__ void k_edge_scatter(const float* __restrict__ v, const float* __restrict__ p,
                               const float* __restrict__ s, const int* __restrict__ src,
                               const int* __restrict__ dst, float* __restrict__ agg, int E) {
  int wave = (blockIdx.x * blockDim.x + threadIdx.x) >> 5;
  int lane = threadIdx.x & 31;
  if (wave >= E * 4) return;
  int e = wave >> 2, h = wave & 3;
  int sn = src[e], dn = dst[e];
  float alpha = p[wave] / s[(long long)dn * 4 + h];
  float4 vv = *(const float4*)(v + (long long)sn * 512 + h * 128 + lane * 4);
  float* ap = agg + (long long)dn * 512 + h * 128 + lane * 4;
  atomicAdd(ap + 0, alpha * vv.x);
  atomicAdd(ap + 1, alpha * vv.y);
  atomicAdd(ap + 2, alpha * vv.z);
  atomicAdd(ap + 3, alpha * vv.w);
}

// ---------------------------------------------------------------------------
// Finalize (float4): head-mean + skip, HeteroConv mean + residual.
// Thread i handles 4 contiguous channels; head stride = 32 float4.
// ---------------------------------------------------------------------------
DEV float4 head_mean4(const float* agg, long long node, int c4) {
  const float4* a = (const float4*)(agg + node * 512) + c4;
  float4 a0 = a[0], a1 = a[32], a2 = a[64], a3 = a[96];
  float4 r;
  r.x = 0.25f * (a0.x + a1.x + a2.x + a3.x);
  r.y = 0.25f * (a0.y + a1.y + a2.y + a3.y);
  r.z = 0.25f * (a0.z + a1.z + a2.z + a3.z);
  r.w = 0.25f * (a0.w + a1.w + a2.w + a3.w);
  return r;
}

__global__ void k_finalize_user(const float* __restrict__ x,
                                const float* __restrict__ aggA, const float* __restrict__ skipA,
                                const float* __restrict__ aggB, const float* __restrict__ skipB,
                                float* __restrict__ out, int nrows) {
  long long i = (long long)blockIdx.x * blockDim.x + threadIdx.x;  // over nrows*32
  if (i >= (long long)nrows * 32) return;
  long long node = i >> 5; int c4 = (int)(i & 31);
  float4 mA = head_mean4(aggA, node, c4);
  float4 mB = head_mean4(aggB, node, c4);
  float4 sA = ((const float4*)skipA)[i];
  float4 sB = ((const float4*)skipB)[i];
  float4 xv = ((const float4*)x)[i];
  float4 o;
  o.x = xv.x + 0.5f * ((mA.x + sA.x) + (mB.x + sB.x));
  o.y = xv.y + 0.5f * ((mA.y + sA.y) + (mB.y + sB.y));
  o.z = xv.z + 0.5f * ((mA.z + sA.z) + (mB.z + sB.z));
  o.w = xv.w + 0.5f * ((mA.w + sA.w) + (mB.w + sB.w));
  ((float4*)out)[i] = o;
}

__global__ void k_finalize_tweet(const float* __restrict__ x,
                                 const float* __restrict__ agg, const float* __restrict__ skip,
                                 float* __restrict__ out, int nrows) {
  long long i = (long long)blockIdx.x * blockDim.x + threadIdx.x;  // over nrows*32
  if (i >= (long long)nrows * 32) return;
  long long node = i >> 5; int c4 = (int)(i & 31);
  float4 mh = head_mean4(agg, node, c4);
  float4 sk = ((const float4*)skip)[i];
  float4 xv = ((const float4*)x)[i];
  float4 o;
  o.x = xv.x + mh.x + sk.x;
  o.y = xv.y + mh.y + sk.y;
  o.z = xv.z + mh.z + sk.z;
  o.w = xv.w + mh.w + sk.w;
  ((float4*)out)[i] = o;
}

// ---------------------------------------------------------------------------
// Host orchestration
// ---------------------------------------------------------------------------
static inline size_t align256(size_t x) { return (x + 255) & ~(size_t)255; }

struct Rel {
  const int *src, *dst; int E;
  const float *Wq, *Wk, *Wv, *Ws, *bq, *bk, *bv, *bs;
  const __bf16 *A_dst, *A_src; int n_dst, n_src;
  __bf16 *pWq, *pWk, *pWv, *pWs;
  float *q, *k, *v, *skip, *logit, *p, *m, *s, *agg;
};

extern "C" void kernel_launch(void* const* d_in, const int* in_sizes, int n_in,
                              void* d_out, int out_size, void* d_ws, size_t ws_size,
                              hipStream_t stream) {
  (void)n_in; (void)out_size; (void)ws_size;
  const float* x_user  = (const float*)d_in[0];
  const float* x_tweet = (const float*)d_in[1];
  const int NUn = in_sizes[0] / 128;
  const int NTn = in_sizes[1] / 128;
  const float* ln_user_g  = (const float*)d_in[8];
  const float* ln_user_b  = (const float*)d_in[9];
  const float* ln_tweet_g = (const float*)d_in[10];
  const float* ln_tweet_b = (const float*)d_in[11];

  char* ws = (char*)d_ws;
  size_t off = 0;
  auto alloc = [&](size_t bytes) -> void* {
    void* ptr = ws + off; off = align256(off + bytes); return ptr;
  };

  __bf16* xu_bf = (__bf16*)alloc((size_t)NUn * 128 * sizeof(__bf16));
  __bf16* xt_bf = (__bf16*)alloc((size_t)NTn * 128 * sizeof(__bf16));

  Rel rel[3];
  const int ndst_tab[3] = {NUn, NTn, NUn};
  const int nsrc_tab[3] = {NUn, NUn, NTn};
  for (int rIdx = 0; rIdx < 3; ++rIdx) {
    Rel& R = rel[rIdx];
    R.src = (const int*)d_in[2 + 2 * rIdx];
    R.dst = (const int*)d_in[3 + 2 * rIdx];
    R.E   = in_sizes[2 + 2 * rIdx];
    int wb = 12 + 8 * rIdx;  // Wq,Wk,Wv,Ws,bq,bk,bv,bs
    R.Wq = (const float*)d_in[wb + 0]; R.Wk = (const float*)d_in[wb + 1];
    R.Wv = (const float*)d_in[wb + 2]; R.Ws = (const float*)d_in[wb + 3];
    R.bq = (const float*)d_in[wb + 4]; R.bk = (const float*)d_in[wb + 5];
    R.bv = (const float*)d_in[wb + 6]; R.bs = (const float*)d_in[wb + 7];
    R.n_dst = ndst_tab[rIdx]; R.n_src = nsrc_tab[rIdx];
    R.A_dst = (rIdx == 1) ? xt_bf : xu_bf;   // post: dst=tweet
    R.A_src = (rIdx == 2) ? xt_bf : xu_bf;   // men:  src=tweet
    R.pWq = (__bf16*)alloc((size_t)128 * 512 * sizeof(__bf16));
    R.pWk = (__bf16*)alloc((size_t)128 * 512 * sizeof(__bf16));
    R.pWv = (__bf16*)alloc((size_t)128 * 512 * sizeof(__bf16));
    R.pWs = (__bf16*)alloc((size_t)128 * 128 * sizeof(__bf16));
    R.q    = (float*)alloc((size_t)R.n_dst * 512 * sizeof(float));
    R.k    = (float*)alloc((size_t)R.n_src * 512 * sizeof(float));
    R.v    = (float*)alloc((size_t)R.n_src * 512 * sizeof(float));
    R.skip = (float*)alloc((size_t)R.n_dst * 128 * sizeof(float));
    R.agg  = (float*)alloc((size_t)R.n_dst * 512 * sizeof(float));
    R.logit = (float*)alloc((size_t)R.E * 4 * sizeof(float));
    R.p     = (float*)alloc((size_t)R.E * 4 * sizeof(float));
    R.m     = (float*)alloc((size_t)R.n_dst * 4 * sizeof(float));
    R.s     = (float*)alloc((size_t)R.n_dst * 4 * sizeof(float));
  }

  const int TB = 256;
  auto blocks1 = [](long long n, int tb) { return (int)((n + tb - 1) / tb); };

  // 1) init m=-inf, s=0, agg=0
  for (int rIdx = 0; rIdx < 3; ++rIdx) {
    Rel& R = rel[rIdx];
    long long nm = (long long)R.n_dst * 4;
    k_fill<<<blocks1(nm, TB), TB, 0, stream>>>(R.m, -__builtin_huge_valf(), nm);
    k_fill<<<blocks1(nm, TB), TB, 0, stream>>>(R.s, 0.0f, nm);
    long long na = (long long)R.n_dst * 512;
    k_fill<<<blocks1(na, TB), TB, 0, stream>>>(R.agg, 0.0f, na);
  }

  // 2) LayerNorm -> bf16
  k_layernorm_bf16<<<blocks1((long long)NUn * 32, TB), TB, 0, stream>>>(
      x_user, ln_user_g, ln_user_b, xu_bf, NUn);
  k_layernorm_bf16<<<blocks1((long long)NTn * 32, TB), TB, 0, stream>>>(
      x_tweet, ln_tweet_g, ln_tweet_b, xt_bf, NTn);

  // 3) pack weights into WMMA B-fragment layout
  for (int rIdx = 0; rIdx < 3; ++rIdx) {
    Rel& R = rel[rIdx];
    k_pack_weight<<<blocks1(128LL * 512, TB), TB, 0, stream>>>(R.Wq, R.pWq, 512);
    k_pack_weight<<<blocks1(128LL * 512, TB), TB, 0, stream>>>(R.Wk, R.pWk, 512);
    k_pack_weight<<<blocks1(128LL * 512, TB), TB, 0, stream>>>(R.Wv, R.pWv, 512);
    k_pack_weight<<<blocks1(128LL * 128, TB), TB, 0, stream>>>(R.Ws, R.pWs, 128);
  }

  // 4) projection GEMMs (WMMA bf16); grid.y = row-tile pairs, waves in x = col tiles
  auto gemm512 = [&](const __bf16* A, const __bf16* Bp, const float* bias, float* C, int M) {
    int mt = M / 16;
    dim3 grid(1, (mt + 1) / 2);
    k_gemm_wmma_bf16<512><<<grid, 256, 0, stream>>>(A, Bp, bias, C, mt);  // 8 waves
  };
  auto gemm128 = [&](const __bf16* A, const __bf16* Bp, const float* bias, float* C, int M) {
    int mt = M / 16;
    dim3 grid(1, (mt + 1) / 2);
    k_gemm_wmma_bf16<128><<<grid, 64, 0, stream>>>(A, Bp, bias, C, mt);   // 2 waves
  };
  for (int rIdx = 0; rIdx < 3; ++rIdx) {
    Rel& R = rel[rIdx];
    gemm512(R.A_dst, R.pWq, R.bq, R.q, R.n_dst);
    gemm512(R.A_src, R.pWk, R.bk, R.k, R.n_src);
    gemm512(R.A_src, R.pWv, R.bv, R.v, R.n_src);
    gemm128(R.A_dst, R.pWs, R.bs, R.skip, R.n_dst);
  }

  // 5) edge attention: logits -> segment max -> exp/sum -> scatter
  for (int rIdx = 0; rIdx < 3; ++rIdx) {
    Rel& R = rel[rIdx];
    k_edge_logits<<<blocks1((long long)R.E, 8), TB, 0, stream>>>(
        R.q, R.k, R.src, R.dst, R.logit, R.E);
    k_edge_max<<<blocks1((long long)R.E * 4, TB), TB, 0, stream>>>(
        R.logit, R.dst, R.m, R.E);
    k_edge_exp<<<blocks1((long long)R.E * 4, TB), TB, 0, stream>>>(
        R.logit, R.dst, R.m, R.p, R.s, R.E);
    k_edge_scatter<<<blocks1((long long)R.E * 4, 8), TB, 0, stream>>>(
        R.v, R.p, R.s, R.src, R.dst, R.agg, R.E);
  }

  // 6) finalize: user gets mean(fol, men), tweet gets post; + residual
  float* out_user  = (float*)d_out;
  float* out_tweet = (float*)d_out + (long long)NUn * 128;
  k_finalize_user<<<blocks1((long long)NUn * 32, TB), TB, 0, stream>>>(
      x_user, rel[0].agg, rel[0].skip, rel[2].agg, rel[2].skip, out_user, NUn);
  k_finalize_tweet<<<blocks1((long long)NTn * 32, TB), TB, 0, stream>>>(
      x_tweet, rel[1].agg, rel[1].skip, out_tweet, NTn);
}
